// MambaPredictor_3633542332583
// MI455X (gfx1250) — matrix operand order repaired and checked by reading, not compiled, not run on hardware
//
#include <hip/hip_runtime.h>
#include <hip/hip_bf16.h>
#include <math.h>

// ---------------- Model constants (match reference) ----------------
#define D_MODEL 768
#define N_LAYERS 2
#define D_STATE 16
#define D_CONV 4
#define D_INNER (2 * D_MODEL)          // 1536
#define DT_RANK ((D_MODEL + 15) / 16)  // 48
#define BATCH 2
#define SEQ 2048
#define NTOK (BATCH * SEQ)             // 4096
#define XDBL_N (DT_RANK + 2 * D_STATE) // 80

typedef __attribute__((ext_vector_type(16))) __bf16 bf16x16;
typedef __attribute__((ext_vector_type(8)))  float  floatx8;

__device__ __forceinline__ float silu_f(float x) {
    return x * (1.0f / (1.0f + __expf(-x)));
}

__device__ __forceinline__ int imin(int a, int b) { return a < b ? a : b; }

// ---------------- simple copy ----------------
__global__ void copy_kernel(const float* __restrict__ src, float* __restrict__ dst, int n) {
    int i = blockIdx.x * blockDim.x + threadIdx.x;
    if (i < n) dst[i] = src[i];
}

// ---------------- LayerNorm: one block per token ----------------
__global__ __launch_bounds__(256) void layernorm_kernel(
    const float* __restrict__ x, const float* __restrict__ w,
    const float* __restrict__ b, float* __restrict__ out, int dim)
{
    const int token = blockIdx.x;
    const float* xr = x + (size_t)token * dim;
    float* orow = out + (size_t)token * dim;

    float s = 0.f, ss = 0.f;
    for (int i = threadIdx.x; i < dim; i += blockDim.x) {
        float v = xr[i];
        s += v; ss += v * v;
    }
    #pragma unroll
    for (int off = 16; off > 0; off >>= 1) {
        s  += __shfl_xor(s,  off, 32);
        ss += __shfl_xor(ss, off, 32);
    }
    __shared__ float sh_s[8], sh_ss[8];
    const int wid = threadIdx.x >> 5, lid = threadIdx.x & 31;
    if (lid == 0) { sh_s[wid] = s; sh_ss[wid] = ss; }
    __syncthreads();
    if (wid == 0) {
        s  = (lid < 8) ? sh_s[lid]  : 0.f;
        ss = (lid < 8) ? sh_ss[lid] : 0.f;
        #pragma unroll
        for (int off = 4; off > 0; off >>= 1) {
            s  += __shfl_xor(s,  off, 32);
            ss += __shfl_xor(ss, off, 32);
        }
        if (lid == 0) { sh_s[0] = s; sh_ss[0] = ss; }
    }
    __syncthreads();
    const float mean = sh_s[0] / (float)dim;
    const float var  = sh_ss[0] / (float)dim - mean * mean;
    const float inv  = rsqrtf(var + 1e-5f);
    for (int i = threadIdx.x; i < dim; i += blockDim.x)
        orow[i] = (xr[i] - mean) * inv * w[i] + b[i];
}

// ---------------- fragment fill helpers ----------------
__device__ __forceinline__ void put4(bf16x16& f, int base, float4 v) {
    f[base + 0] = (__bf16)v.x;
    f[base + 1] = (__bf16)v.y;
    f[base + 2] = (__bf16)v.z;
    f[base + 3] = (__bf16)v.w;
}

// ---------------- WMMA GEMM: C[m][n] = sum_k A[m][k]*W[n][k] (+bias)(+=C) ----------------
// Block = 128 threads = 4 waves. Wave computes a 32x64 output tile:
// 2 M-subtiles x 4 N-subtiles = 8 independent WMMA chains per K-step
// (B fragments reused across both M-subtiles; A fragments reused across 4 N-subtiles).
// Unconditional vectorized loads via row clamping; K-tail handled branchlessly.
template <bool BIAS, bool ACC>
__global__ __launch_bounds__(128) void wmma_gemm_kernel(
    const float* __restrict__ A, int lda,
    const float* __restrict__ W,            // [N][K] row-major, row stride K
    float* __restrict__ C, int ldc,
    const float* __restrict__ bias,         // length N (used iff BIAS)
    int M, int N, int K)
{
    const int wave = threadIdx.x >> 5;
    const int lane = threadIdx.x & 31;
    const int g = lane >> 4;   // half-wave group
    const int r = lane & 15;

    const int mtile = blockIdx.y * 128 + wave * 32;
    const int ntile = blockIdx.x * 64;

    floatx8 acc[8] = {};   // acc[mt*4 + nt]

    // Clamp rows so loads are always in-range; OOB results are masked at store.
    const float* Arow0 = A + (size_t)imin(mtile +  0 + r, M - 1) * lda;
    const float* Arow1 = A + (size_t)imin(mtile + 16 + r, M - 1) * lda;
    const float* Arows[2] = { Arow0, Arow1 };
    const float* Wrow0 = W + (size_t)imin(ntile +  0 + r, N - 1) * K;
    const float* Wrow1 = W + (size_t)imin(ntile + 16 + r, N - 1) * K;
    const float* Wrow2 = W + (size_t)imin(ntile + 32 + r, N - 1) * K;
    const float* Wrow3 = W + (size_t)imin(ntile + 48 + r, N - 1) * K;
    const float* Wrows[4] = { Wrow0, Wrow1, Wrow2, Wrow3 };

    const int Kfull = K & ~31;

    for (int kb = 0; kb < Kfull; kb += 32) {
        // A fragments (16x32 bf16): lane holds one row.
        //   e=0..7  -> k = kb + g*8 + e        (contiguous 8 floats)
        //   e=8..15 -> k = kb + 16 + g*8 + e-8 (contiguous 8 floats)
        bf16x16 afrag[2];
        #pragma unroll
        for (int mt = 0; mt < 2; mt++) {
            const float* ab = Arows[mt] + kb + g * 8;
            put4(afrag[mt], 0,  *(const float4*)(ab));
            put4(afrag[mt], 4,  *(const float4*)(ab + 4));
            put4(afrag[mt], 8,  *(const float4*)(ab + 16));
            put4(afrag[mt], 12, *(const float4*)(ab + 20));
        }
        #pragma unroll
        for (int nt = 0; nt < 4; nt++) {
            // B fragment (32x16 bf16): lane holds one column; same k order as A
            const float* bb = Wrows[nt] + kb + g * 8;
            bf16x16 bfrag;
            put4(bfrag, 0,  *(const float4*)(bb));
            put4(bfrag, 4,  *(const float4*)(bb + 4));
            put4(bfrag, 8,  *(const float4*)(bb + 16));
            put4(bfrag, 12, *(const float4*)(bb + 20));
            #pragma unroll
            for (int mt = 0; mt < 2; mt++) {
                acc[mt * 4 + nt] = __builtin_amdgcn_wmma_f32_16x16x32_bf16(
                    false, afrag[mt], false, bfrag, (short)0, acc[mt * 4 + nt], false, false);
            }
        }
    }

    // K tail (only dt_proj: K=48). Clamped addresses + branchless zero-select.
    if (Kfull < K) {
        const int kb = Kfull;
        bf16x16 afrag[2];
        #pragma unroll
        for (int mt = 0; mt < 2; mt++) {
            #pragma unroll
            for (int e = 0; e < 16; e++) {
                int k = kb + g * 8 + (e & 7) + ((e & 8) ? 16 : 0);
                float v = Arows[mt][imin(k, K - 1)];
                afrag[mt][e] = (__bf16)((k < K) ? v : 0.f);
            }
        }
        #pragma unroll
        for (int nt = 0; nt < 4; nt++) {
            bf16x16 bfrag;
            #pragma unroll
            for (int e = 0; e < 16; e++) {
                int k = kb + g * 8 + (e & 7) + ((e & 8) ? 16 : 0);
                float v = Wrows[nt][imin(k, K - 1)];
                bfrag[e] = (__bf16)((k < K) ? v : 0.f);
            }
            #pragma unroll
            for (int mt = 0; mt < 2; mt++) {
                acc[mt * 4 + nt] = __builtin_amdgcn_wmma_f32_16x16x32_bf16(
                    false, afrag[mt], false, bfrag, (short)0, acc[mt * 4 + nt], false, false);
            }
        }
    }

    // Store: VGPR j holds row subtile_m + g*8 + j, col = ntile + nt*16 + (lane&15)
    #pragma unroll
    for (int nt = 0; nt < 4; nt++) {
        const int n = ntile + nt * 16 + r;
        if (n < N) {
            const float bv = BIAS ? bias[n] : 0.f;
            #pragma unroll
            for (int mt = 0; mt < 2; mt++) {
                #pragma unroll
                for (int j = 0; j < 8; j++) {
                    const int m = mtile + mt * 16 + g * 8 + j;
                    if (m < M) {
                        float v = acc[mt * 4 + nt][j] + bv;
                        if (ACC) v += C[(size_t)m * ldc + n];
                        C[(size_t)m * ldc + n] = v;
                    }
                }
            }
        }
    }
}

// ---------------- depthwise causal conv (D_CONV=4) + SiLU ----------------
__global__ void conv_silu_kernel(const float* __restrict__ xz,
                                 const float* __restrict__ cw,   // [D_INNER][D_CONV]
                                 const float* __restrict__ cb,   // [D_INNER]
                                 float* __restrict__ u, int total)
{
    int idx = blockIdx.x * blockDim.x + threadIdx.x;
    if (idx >= total) return;
    const int d = idx % D_INNER;
    const int t = (idx / D_INNER) % SEQ;
    const int b = idx / (D_INNER * SEQ);
    const float* wrow = cw + (size_t)d * D_CONV;
    float acc = cb[d];
    #pragma unroll
    for (int j = 0; j < D_CONV; j++) {
        int tt = t - (D_CONV - 1) + j;
        if (tt >= 0)
            acc += wrow[j] * xz[((size_t)b * SEQ + tt) * (2 * D_INNER) + d];
    }
    u[idx] = silu_f(acc);
}

// ---------------- dt = softplus(dt_lin + bias) ----------------
__global__ void softplus_bias_kernel(float* __restrict__ dt,
                                     const float* __restrict__ bias,
                                     int total)
{
    int idx = blockIdx.x * blockDim.x + threadIdx.x;
    if (idx >= total) return;
    float x = dt[idx] + bias[idx % D_INNER];
    dt[idx] = (x > 20.f) ? x : log1pf(__expf(x));
}

// ---------------- selective scan: one thread per (batch, channel) ----------------
// Block = 256 channels of the same batch; B/C row staged in LDS each step.
__global__ __launch_bounds__(256) void scan_kernel(
    const float* __restrict__ u,     // [NTOK][D_INNER]
    const float* __restrict__ dt,    // [NTOK][D_INNER]
    const float* __restrict__ xdbl,  // [NTOK][80]; B at [48..64), C at [64..80)
    const float* __restrict__ A_log, // [D_INNER][D_STATE]
    const float* __restrict__ Dp,    // [D_INNER]
    float* __restrict__ y)           // [NTOK][D_INNER]
{
    const int idx = blockIdx.x * blockDim.x + threadIdx.x; // < BATCH*D_INNER (=3072, exact)
    const int d = idx % D_INNER;
    const int b = idx / D_INNER;

    float A[D_STATE], h[D_STATE];
    #pragma unroll
    for (int s = 0; s < D_STATE; s++) {
        A[s] = -__expf(A_log[(size_t)d * D_STATE + s]);
        h[s] = 0.f;
    }
    const float Dd = Dp[d];

    __shared__ float sBC[2 * D_STATE];

    for (int t = 0; t < SEQ; t++) {
        const size_t row = (size_t)b * SEQ + t;
        if (threadIdx.x < 2 * D_STATE)
            sBC[threadIdx.x] = xdbl[row * XDBL_N + DT_RANK + threadIdx.x];
        __syncthreads();

        const float dtv = dt[row * D_INNER + d];
        const float uv  = u[row * D_INNER + d];
        const float du  = dtv * uv;
        float acc = 0.f;
        #pragma unroll
        for (int s = 0; s < D_STATE; s++) {
            float hb = __expf(dtv * A[s]) * h[s] + du * sBC[s];
            h[s] = hb;
            acc += hb * sBC[D_STATE + s];
        }
        y[row * D_INNER + d] = acc + uv * Dd;
        __syncthreads();
    }
}

// ---------------- gate: y *= silu(z), z = xz[..., D_INNER + d] ----------------
__global__ void gate_kernel(float* __restrict__ y, const float* __restrict__ xz, int total)
{
    int idx = blockIdx.x * blockDim.x + threadIdx.x;
    if (idx >= total) return;
    const int d = idx % D_INNER;
    const size_t row = (size_t)idx / D_INNER;
    const float z = xz[row * (2 * D_INNER) + D_INNER + d];
    y[idx] *= silu_f(z);
}

// ---------------- host driver ----------------
extern "C" void kernel_launch(void* const* d_in, const int* in_sizes, int n_in,
                              void* d_out, int out_size, void* d_ws, size_t ws_size,
                              hipStream_t stream)
{
    const float* x         = (const float*)d_in[0];
    const float* in_proj_w = (const float*)d_in[1];
    const float* conv_w    = (const float*)d_in[2];
    const float* conv_b    = (const float*)d_in[3];
    const float* x_proj_w  = (const float*)d_in[4];
    const float* dt_proj_w = (const float*)d_in[5];
    const float* dt_proj_b = (const float*)d_in[6];
    const float* A_log     = (const float*)d_in[7];
    const float* D_param   = (const float*)d_in[8];
    const float* out_proj_w= (const float*)d_in[9];
    const float* ln_w      = (const float*)d_in[10];
    const float* ln_b      = (const float*)d_in[11];
    const float* fnorm_w   = (const float*)d_in[12];
    const float* fnorm_b   = (const float*)d_in[13];
    const float* proj_w    = (const float*)d_in[14];
    const float* proj_b    = (const float*)d_in[15];
    float* out = (float*)d_out;

    float* ws = (float*)d_ws;
    float* h    = ws;                                 // NTOK*D_MODEL residual stream
    float* hln  = h    + (size_t)NTOK * D_MODEL;      // NTOK*D_MODEL
    float* xz   = hln  + (size_t)NTOK * D_MODEL;      // NTOK*2*D_INNER
    float* uact = xz   + (size_t)NTOK * 2 * D_INNER;  // NTOK*D_INNER
    float* xdbl = uact + (size_t)NTOK * D_INNER;      // NTOK*80
    float* dtb  = xdbl + (size_t)NTOK * XDBL_N;       // NTOK*D_INNER
    float* yb   = dtb  + (size_t)NTOK * D_INNER;      // NTOK*D_INNER

    const int nElemD = NTOK * D_MODEL;   // 3,145,728
    const int nElemI = NTOK * D_INNER;   // 6,291,456

    // residual stream h = x
    copy_kernel<<<(nElemD + 255) / 256, 256, 0, stream>>>(x, h, nElemD);

    for (int i = 0; i < N_LAYERS; i++) {
        // 1) LN
        layernorm_kernel<<<NTOK, 256, 0, stream>>>(
            h, ln_w + (size_t)i * D_MODEL, ln_b + (size_t)i * D_MODEL, hln, D_MODEL);

        // 2) in_proj: xz = hln @ W^T   (M=4096, K=768, N=3072)
        wmma_gemm_kernel<false, false><<<dim3((2 * D_INNER) / 64, NTOK / 128), 128, 0, stream>>>(
            hln, D_MODEL,
            in_proj_w + (size_t)i * 2 * D_INNER * D_MODEL,
            xz, 2 * D_INNER, nullptr,
            NTOK, 2 * D_INNER, D_MODEL);

        // 3) depthwise causal conv + SiLU
        conv_silu_kernel<<<(nElemI + 255) / 256, 256, 0, stream>>>(
            xz, conv_w + (size_t)i * D_INNER * D_CONV, conv_b + (size_t)i * D_INNER,
            uact, nElemI);

        // 4) x_proj: xdbl = uact @ W^T  (M=4096, K=1536, N=80)
        wmma_gemm_kernel<false, false><<<dim3((XDBL_N + 63) / 64, NTOK / 128), 128, 0, stream>>>(
            uact, D_INNER,
            x_proj_w + (size_t)i * XDBL_N * D_INNER,
            xdbl, XDBL_N, nullptr,
            NTOK, XDBL_N, D_INNER);

        // 5) dt_proj: dtb = xdbl[:, :48] @ W^T  (M=4096, K=48 (lda=80), N=1536)
        wmma_gemm_kernel<false, false><<<dim3(D_INNER / 64, NTOK / 128), 128, 0, stream>>>(
            xdbl, XDBL_N,
            dt_proj_w + (size_t)i * D_INNER * DT_RANK,
            dtb, D_INNER, nullptr,
            NTOK, D_INNER, DT_RANK);
        softplus_bias_kernel<<<(nElemI + 255) / 256, 256, 0, stream>>>(
            dtb, dt_proj_b + (size_t)i * D_INNER, nElemI);

        // 6) selective scan
        scan_kernel<<<(BATCH * D_INNER) / 256, 256, 0, stream>>>(
            uact, dtb, xdbl,
            A_log + (size_t)i * D_INNER * D_STATE,
            D_param + (size_t)i * D_INNER, yb);

        // 7) gate y *= silu(z)
        gate_kernel<<<(nElemI + 255) / 256, 256, 0, stream>>>(yb, xz, nElemI);

        // 8) out_proj with fused residual: h += yb @ W^T  (M=4096, K=1536, N=768)
        wmma_gemm_kernel<false, true><<<dim3(D_MODEL / 64, NTOK / 128), 128, 0, stream>>>(
            yb, D_INNER,
            out_proj_w + (size_t)i * D_MODEL * D_INNER,
            h, D_MODEL, nullptr,
            NTOK, D_MODEL, D_INNER);
    }

    // final LN + projection (+bias): out = LN(h) @ proj_w^T + proj_b
    layernorm_kernel<<<NTOK, 256, 0, stream>>>(h, fnorm_w, fnorm_b, hln, D_MODEL);
    wmma_gemm_kernel<true, false><<<dim3(D_MODEL / 64, NTOK / 128), 128, 0, stream>>>(
        hln, D_MODEL, proj_w, out, D_MODEL, proj_b,
        NTOK, D_MODEL, D_MODEL);
}
